// SketchMem_35974646071383
// MI455X (gfx1250) — matrix operand order repaired and checked by reading, not compiled, not run on hardware
//
#include <hip/hip_runtime.h>

// =====================================================================
// Fused memory-network layer for gfx1250 (MI455X), bf16 WMMA pipeline.
//
//  out = concat( softmax(l2n(QWb^T+bb) @ l2n(Kb)^T) @ Vb ,
//                softmax((QWc^T+bc) . Kc[idx]) @ Vc[idx] ) @ Wo^T + bo
//
// HBM-bound: read Q once (134MB), write out once (134MB). Weights are
// pre-swizzled into WMMA B-fragment layout in d_ws; the hottest bank
// (fused projection weights) is staged in LDS per block and read with a
// one-deep software pipeline so ds latency hides under WMMAs.
// =====================================================================

typedef __attribute__((ext_vector_type(16))) __bf16 v16bf;
typedef __attribute__((ext_vector_type(8)))  __bf16 v8bf;
typedef __attribute__((ext_vector_type(4)))  __bf16 v4bf;
typedef __attribute__((ext_vector_type(8)))  float  v8f;
typedef __attribute__((ext_vector_type(4)))  float  f4;
typedef __attribute__((ext_vector_type(4)))  int    i4;

// d_ws layout (units: bf16 halves)
#define WBC_H 0        // [512,128] fused Wbase|Wcat^T frags : 16 t * 8 j  * 512
#define WO_H  65536    // [128,512] Wout^T frags             : 4 t  * 32 j * 512
#define NK_H  131072   // [64,112]  l2norm(base_keys)^T frags: 2 t  * 7 j  * 512
#define BV_H  138240   // [128,64]  base_vals (zero-pad) frags:4 t  * 4 j  * 512
// total halves = 146432 -> 292864 bytes of d_ws needed

__device__ __forceinline__ v8f wmma_bf16(v16bf a, v16bf b, v8f c) {
  return __builtin_amdgcn_wmma_f32_16x16x32_bf16(false, a, false, b, (short)0, c,
                                                 false, false);
}

__device__ __forceinline__ v16bf cat8(v8bf lo, v8bf hi) {
  return __builtin_shufflevector(lo, hi, 0,1,2,3,4,5,6,7,8,9,10,11,12,13,14,15);
}
__device__ __forceinline__ v8bf cat4(v4bf lo, v4bf hi) {
  return __builtin_shufflevector(lo, hi, 0,1,2,3,4,5,6,7);
}
__device__ __forceinline__ v4bf cvt4(f4 x) {
  return __builtin_convertvector(x, v4bf);   // packed f32->bf16 (RNE)
}
__device__ __forceinline__ v8f splat8(float x) {
  return (v8f){x,x,x,x,x,x,x,x};
}

// A-fragment (16x32 bf16) gather from a row-major [16 x stride] bf16 array.
// Lane: m = lane&15, hi = lane>>4. Halves 0..7 = cols kbase+hi*8+0..7,
// halves 8..15 = cols kbase+16+hi*8+0..7 (per CDNA5 ISA 7.12.2).
__device__ __forceinline__ v16bf lds_afrag(const __bf16* base) {
  v8bf lo = *(const v8bf*)(base);
  v8bf hi = *(const v8bf*)(base + 16);
  return cat8(lo, hi);
}

// ---------------------------------------------------------------------
// Prep: build swizzled B-fragments in d_ws. B-frag layout (16-bit B,
// 32x16): lane L holds column n = j*16+(L&15); half h holds
// K = t*32 + (h>>3)*16 + (L>>4)*8 + (h&7). Fragment storage: ((t*NT+j)*32+L)*16+h.
// ---------------------------------------------------------------------
__global__ __launch_bounds__(256) void prep_kernel(
    const float* __restrict__ W_base, const float* __restrict__ W_cat,
    const float* __restrict__ base_keys, const float* __restrict__ base_vals,
    const float* __restrict__ W_out, __bf16* __restrict__ ws) {
  const int tid = blockIdx.x * blockDim.x + threadIdx.x;
  const int nth = gridDim.x * blockDim.x;

  // Region 1: fused [W_base;W_cat]^T  (K=512 -> 16 tsteps, N=128 -> 8 tiles)
  for (int i = tid; i < 16 * 8 * 512; i += nth) {
    int h = i & 15, L = (i >> 4) & 31, tj = i >> 9;
    int t = tj >> 3, j = tj & 7;
    int k = t * 32 + (h >> 3) * 16 + (L >> 4) * 8 + (h & 7);
    int n = j * 16 + (L & 15);
    float v = (n < 64) ? W_base[n * 512 + k] : W_cat[(n - 64) * 512 + k];
    ws[WBC_H + i] = (__bf16)v;
  }
  // Region 2: W_out^T (K=128 -> 4 tsteps, N=512 -> 32 tiles)
  for (int i = tid; i < 4 * 32 * 512; i += nth) {
    int h = i & 15, L = (i >> 4) & 31, tj = i >> 9;
    int t = tj >> 5, j = tj & 31;
    int k = t * 32 + (h >> 3) * 16 + (L >> 4) * 8 + (h & 7);
    int n = j * 16 + (L & 15);
    ws[WO_H + i] = (__bf16)W_out[n * 128 + k];
  }
  // Region 3: l2norm(base_keys)^T (K=64 -> 2 tsteps, N=112 pad -> 7 tiles)
  for (int i = tid; i < 2 * 7 * 512; i += nth) {
    int h = i & 15, L = (i >> 4) & 31, tj = i >> 9;
    int t = tj / 7, j = tj % 7;
    int k = t * 32 + (h >> 3) * 16 + (L >> 4) * 8 + (h & 7);
    int n = j * 16 + (L & 15);
    float v = 0.f;
    if (n < 100) {
      float ss = 0.f;
      for (int d = 0; d < 64; ++d) { float x = base_keys[n * 64 + d]; ss += x * x; }
      v = base_keys[n * 64 + k] / fmaxf(sqrtf(ss), 1e-8f);
    }
    ws[NK_H + i] = (__bf16)v;
  }
  // Region 4: base_vals zero-padded to K=128 (4 tsteps, N=64 -> 4 tiles)
  for (int i = tid; i < 4 * 4 * 512; i += nth) {
    int h = i & 15, L = (i >> 4) & 31, tj = i >> 9;
    int t = tj >> 2, j = tj & 3;
    int k = t * 32 + (h >> 3) * 16 + (L >> 4) * 8 + (h & 7);
    int n = j * 16 + (L & 15);
    ws[BV_H + i] = (__bf16)((k < 100) ? base_vals[k * 64 + n] : 0.f);
  }
}

// ---------------------------------------------------------------------
// Fused kernel. 8 waves/block, each wave owns 16 rows end-to-end.
// ---------------------------------------------------------------------
__global__ __launch_bounds__(256) void fused_kernel(
    const float* __restrict__ q, const int* __restrict__ cat_idx,
    const float* __restrict__ b_base, const float* __restrict__ b_cat,
    const float* __restrict__ cat_keys, const float* __restrict__ cat_vals,
    const float* __restrict__ b_out, const __bf16* __restrict__ ws,
    float* __restrict__ out) {
  __shared__ __bf16 lwbc[16 * 8 * 512];  // 128KB: WBC fragments (hot loop)
  __shared__ __bf16 smem[8 * 2048];      // 4KB per-wave scratch

  const int tid  = threadIdx.x;
  const int wid  = tid >> 5;
  const int lane = tid & 31;
  const int ln15 = lane & 15;
  const int hi   = lane >> 4;
  const int rowBase = blockIdx.x * 128 + wid * 16;

  // ---- cooperative stage of WBC fragments into LDS (b128 copies) ----
  {
    const i4* src = (const i4*)(ws + WBC_H);
    i4* dst = (i4*)lwbc;
#pragma unroll
    for (int c = 0; c < 32; ++c)
      dst[tid + c * 256] = src[tid + c * 256];
  }
  __syncthreads();

  __bf16* sm  = smem + wid * 2048;  // bf16 view
  float*  smf = (float*)sm;         // f32 view ([16][64])

  // ===== Step A: [16x128] = Q[16x512] @ [Wb|Wc]^T, bf16 WMMA, K=512 =====
  // Bias is column-broadcast in the C layout -> fold into accumulator init.
  v8f accA[8];
#pragma unroll
  for (int j = 0; j < 8; ++j) {
    float bias = (j < 4) ? b_base[j * 16 + ln15] : b_cat[(j - 4) * 16 + ln15];
    accA[j] = splat8(bias);
  }

  const float* qrow = q + (size_t)(rowBase + ln15) * 512 + hi * 8;
  const __bf16* wbbase = lwbc + lane * 16;
  // one-deep pipeline: fragment for (t,j) is loaded during WMMA (t,j-1)
  v16bf bcur = *(const v16bf*)(wbbase);
#pragma unroll
  for (int t = 0; t < 16; ++t) {
    const float* qp = qrow + t * 32;
    f4 a0 = __builtin_nontemporal_load((const f4*)(qp));
    f4 a1 = __builtin_nontemporal_load((const f4*)(qp + 4));
    f4 a2 = __builtin_nontemporal_load((const f4*)(qp + 16));
    f4 a3 = __builtin_nontemporal_load((const f4*)(qp + 20));
    v16bf af = cat8(cat4(cvt4(a0), cvt4(a1)), cat4(cvt4(a2), cvt4(a3)));
#pragma unroll
    for (int j = 0; j < 8; ++j) {
      const int nidx = t * 8 + j + 1;
      v16bf bnext = *(const v16bf*)(wbbase + (nidx < 128 ? nidx : 0) * 512);
      accA[j] = wmma_bf16(af, bcur, accA[j]);
      bcur = bnext;
    }
  }

  // ===== stage cat_query f32 -> scratch [16][64] =====
#pragma unroll
  for (int j = 4; j < 8; ++j)
#pragma unroll
    for (int r = 0; r < 8; ++r)
      smf[(r + 8 * hi) * 64 + (j - 4) * 16 + ln15] = accA[j][r];

  // ===== category path (VALU, 2 lanes/row) =====
  const int crow  = lane >> 1;
  const int chalf = lane & 1;
  const int ci    = cat_idx[rowBase + crow];
  const float* ckp = cat_keys + (size_t)ci * 5 * 64 + chalf * 32;
  const float* cvp = cat_vals + (size_t)ci * 5 * 64 + chalf * 32;
  float csims[5];
#pragma unroll
  for (int kk = 0; kk < 5; ++kk) {
    const f4* kr = (const f4*)(ckp + kk * 64);
    const f4* qr = (const f4*)(smf + crow * 64 + chalf * 32);
    float s = 0.f;
#pragma unroll
    for (int d = 0; d < 8; ++d) {
      f4 a = qr[d], b = kr[d];
      s += a[0]*b[0] + a[1]*b[1] + a[2]*b[2] + a[3]*b[3];
    }
    s += __shfl_xor(s, 1, 32);
    csims[kk] = s;
  }
  float cm = csims[0];
#pragma unroll
  for (int kk = 1; kk < 5; ++kk) cm = fmaxf(cm, csims[kk]);
  float ce[5], cs = 0.f;
#pragma unroll
  for (int kk = 0; kk < 5; ++kk) { ce[kk] = __expf(csims[kk] - cm); cs += ce[kk]; }
  float cinv = 1.f / cs;
  f4 clook[8];
#pragma unroll
  for (int d = 0; d < 8; ++d) clook[d] = (f4){0.f, 0.f, 0.f, 0.f};
#pragma unroll
  for (int kk = 0; kk < 5; ++kk) {
    float a = ce[kk] * cinv;
    const f4* vr = (const f4*)(cvp + kk * 64);
#pragma unroll
    for (int d = 0; d < 8; ++d) {
      f4 v = vr[d];
      clook[d] += (f4){a*v[0], a*v[1], a*v[2], a*v[3]};
    }
  }

  // ===== base path: l2-normalize rows of base_query =====
  float scale[8];
#pragma unroll
  for (int r = 0; r < 8; ++r) {
    float s = 0.f;
#pragma unroll
    for (int j = 0; j < 4; ++j) s += accA[j][r] * accA[j][r];
#pragma unroll
    for (int off = 1; off < 16; off <<= 1) s += __shfl_xor(s, off, 32);
    scale[r] = 1.f / fmaxf(sqrtf(s), 1e-8f);
  }
  // stage normalized base_query bf16 -> scratch [16][64]
#pragma unroll
  for (int j = 0; j < 4; ++j)
#pragma unroll
    for (int r = 0; r < 8; ++r)
      sm[(r + 8 * hi) * 64 + j * 16 + ln15] = (__bf16)(accA[j][r] * scale[r]);

  // ===== sims GEMM: [16x112] = nbq[16x64] @ nkeys^T, K=64 =====
  v8f accS[7];
#pragma unroll
  for (int j = 0; j < 7; ++j) accS[j] = splat8(0.f);
#pragma unroll
  for (int t = 0; t < 2; ++t) {
    v16bf af = lds_afrag(sm + ln15 * 64 + t * 32 + hi * 8);
    const __bf16* nk = ws + NK_H + t * 3584 + lane * 16;
    v16bf nkf[7];
#pragma unroll
    for (int j = 0; j < 7; ++j) nkf[j] = *(const v16bf*)(nk + j * 512);
#pragma unroll
    for (int j = 0; j < 7; ++j) accS[j] = wmma_bf16(af, nkf[j], accS[j]);
  }
  // mask pad cols 100..111 (tile 6, ln15>=4), softmax over N, stage attn bf16
  if (ln15 >= 4) {
#pragma unroll
    for (int r = 0; r < 8; ++r) accS[6][r] = -1e30f;
  }
#pragma unroll
  for (int r = 0; r < 8; ++r) {
    float m = accS[0][r];
#pragma unroll
    for (int j = 1; j < 7; ++j) m = fmaxf(m, accS[j][r]);
#pragma unroll
    for (int off = 1; off < 16; off <<= 1) m = fmaxf(m, __shfl_xor(m, off, 32));
    float e[7], s = 0.f;
#pragma unroll
    for (int j = 0; j < 7; ++j) { e[j] = __expf(accS[j][r] - m); s += e[j]; }
#pragma unroll
    for (int off = 1; off < 16; off <<= 1) s += __shfl_xor(s, off, 32);
    float inv = 1.f / s;
#pragma unroll
    for (int j = 0; j < 7; ++j)
      sm[(r + 8 * hi) * 128 + j * 16 + ln15] = (__bf16)(e[j] * inv);
    sm[(r + 8 * hi) * 128 + 112 + ln15] = (__bf16)0.f;  // zero pad cols 112..127
  }

  // ===== base lookup GEMM: [16x64] = attn[16x128] @ base_vals, K=128 =====
  v8f accL[4];
#pragma unroll
  for (int j = 0; j < 4; ++j) accL[j] = splat8(0.f);
#pragma unroll
  for (int t = 0; t < 4; ++t) {
    v16bf af = lds_afrag(sm + ln15 * 128 + t * 32 + hi * 8);
    const __bf16* bv = ws + BV_H + t * 2048 + lane * 16;
    v16bf bvf[4];
#pragma unroll
    for (int j = 0; j < 4; ++j) bvf[j] = *(const v16bf*)(bv + j * 512);
#pragma unroll
    for (int j = 0; j < 4; ++j) accL[j] = wmma_bf16(af, bvf[j], accL[j]);
  }

  // ===== stage combined [16][128] bf16 = base_lookup | cat_lookup =====
#pragma unroll
  for (int j = 0; j < 4; ++j)
#pragma unroll
    for (int r = 0; r < 8; ++r)
      sm[(r + 8 * hi) * 128 + j * 16 + ln15] = (__bf16)(accL[j][r]);
#pragma unroll
  for (int d = 0; d < 8; ++d) {
    v4bf cv = cvt4(clook[d]);
    *(v4bf*)(sm + crow * 128 + 64 + chalf * 32 + d * 4) = cv;
  }

  // ===== output GEMM: [16x512] = combined[16x128] @ Wout^T, K=128 =====
  v16bf cf[4];
#pragma unroll
  for (int t = 0; t < 4; ++t)
    cf[t] = lds_afrag(sm + ln15 * 128 + t * 32 + hi * 8);

  // double-buffered W_out fragments: next tile's loads fly over this
  // tile's WMMAs
  const __bf16* wo0 = ws + WO_H + lane * 16;
  v16bf wof[2][4];
#pragma unroll
  for (int t = 0; t < 4; ++t)
    wof[0][t] = *(const v16bf*)(wo0 + t * 16384);

  for (int jg = 0; jg < 32; ++jg) {
    const int cur = jg & 1, nxt = cur ^ 1;
    if (jg < 31) {
      const __bf16* won = wo0 + (jg + 1) * 512;
#pragma unroll
      for (int t = 0; t < 4; ++t)
        wof[nxt][t] = *(const v16bf*)(won + t * 16384);
    }
    // b_out is column-broadcast in C layout -> fold into accumulator init
    v8f acc = splat8(b_out[jg * 16 + ln15]);
#pragma unroll
    for (int t = 0; t < 4; ++t)
      acc = wmma_bf16(cf[t], wof[cur][t], acc);
    float* op = out + (size_t)(rowBase + 8 * hi) * 512 + jg * 16 + ln15;
#pragma unroll
    for (int r = 0; r < 8; ++r)
      __builtin_nontemporal_store(acc[r], op + (size_t)r * 512);
  }
}

extern "C" void kernel_launch(void* const* d_in, const int* in_sizes, int n_in,
                              void* d_out, int out_size, void* d_ws, size_t ws_size,
                              hipStream_t stream) {
  (void)in_sizes; (void)n_in; (void)out_size; (void)ws_size;
  const float* query     = (const float*)d_in[0];
  const int*   cat_idx   = (const int*)d_in[1];
  const float* W_base    = (const float*)d_in[2];
  const float* b_base    = (const float*)d_in[3];
  const float* W_cat     = (const float*)d_in[4];
  const float* b_cat     = (const float*)d_in[5];
  const float* base_keys = (const float*)d_in[6];
  const float* base_vals = (const float*)d_in[7];
  const float* cat_keys  = (const float*)d_in[8];
  const float* cat_vals  = (const float*)d_in[9];
  const float* W_out     = (const float*)d_in[10];
  const float* b_out     = (const float*)d_in[11];
  __bf16* ws = (__bf16*)d_ws;  // needs >= 292864 bytes

  prep_kernel<<<144, 256, 0, stream>>>(W_base, W_cat, base_keys, base_vals, W_out, ws);
  fused_kernel<<<512, 256, 0, stream>>>(query, cat_idx, b_base, b_cat,
                                        cat_keys, cat_vals, b_out, ws,
                                        (float*)d_out);
}
